// GCNNet_3375844295345
// MI455X (gfx1250) — matrix operand order repaired and checked by reading, not compiled, not run on hardware
//
#include <hip/hip_runtime.h>
#include <hip/hip_bf16.h>
#include <cstdint>

#define F_IN 512
#define HID  16
#define NCLS 40
#define KSTEPS (F_IN / 32)   // 16 WMMA K-steps

typedef __attribute__((ext_vector_type(16))) __bf16 v16bf;
typedef __attribute__((ext_vector_type(8)))  float  v8f;
typedef __attribute__((ext_vector_type(4)))  float  v4f;

// ---------------- degree / normalization ----------------
__global__ __launch_bounds__(256) void k_deg_init(float* __restrict__ deg, int n) {
  int i = blockIdx.x * 256 + threadIdx.x;
  if (i < n) deg[i] = 1.0f;                       // self-loop contribution
}

__global__ __launch_bounds__(256) void k_deg_edges(const int* __restrict__ col,
                                                   float* __restrict__ deg, int e) {
  int i = blockIdx.x * 256 + threadIdx.x;
  if (i < e) atomicAdd(&deg[col[i]], 1.0f);
}

__global__ __launch_bounds__(256) void k_dinv(const float* __restrict__ deg,
                                              float* __restrict__ dinv, int n) {
  int i = blockIdx.x * 256 + threadIdx.x;
  if (i < n) dinv[i] = rsqrtf(deg[i]);            // deg >= 1 always
}

// ---------------- layer-1 GEMM: h = x @ W1 via bf16 WMMA ----------------
// One wave per 16-row tile. W1 is pre-swizzled into LDS in the exact per-lane
// B-fragment layout, so each K-step's B operand is one 32B contiguous LDS read
// (2x ds_load_b128). A operand is 4x global_load_b128 + cvt_pk_bf16.
__global__ __launch_bounds__(256) void k_gemm1_wmma(const float* __restrict__ x,
                                                    const float* __restrict__ W1,
                                                    float* __restrict__ h,
                                                    int nrows, int ntiles) {
  __shared__ v16bf bfrag[KSTEPS * 32];            // 16 steps * 32 lanes * 32B = 16 KB
  {
    __bf16* ws = (__bf16*)bfrag;
    // element j = ((step*32 + lane)*16 + i) holds W1[(step*32 + 8*(lane>>4) + i + (i&8))*HID + (lane&15)]
    for (int j = threadIdx.x; j < KSTEPS * 32 * 16; j += 256) {
      int i    = j & 15;
      int lane = (j >> 4) & 31;
      int step = j >> 9;
      int k    = step * 32 + 8 * (lane >> 4) + i + (i & 8);
      ws[j] = (__bf16)W1[k * HID + (lane & 15)];
    }
  }
  __syncthreads();

  const int wave = threadIdx.x >> 5;
  const int lane = threadIdx.x & 31;
  const int half = lane >> 4;
  const int m16  = lane & 15;
  const int tile = blockIdx.x * 8 + wave;
  if (tile >= ntiles) return;                     // wave-uniform: EXEC all-1s for WMMA
  const int row0 = tile << 4;

  int arow = row0 + m16; if (arow >= nrows) arow = nrows - 1;
  const float* __restrict__ xr = x + (size_t)arow * F_IN + 8 * half;

  v8f acc = {};
#pragma unroll 4
  for (int step = 0; step < KSTEPS; ++step) {
    // A (16x32 bf16): elems 0..7 -> K = kk+8*half+{0..7}; elems 8..15 -> +16
    const v4f* xv = (const v4f*)(xr + step * 32);
    v4f p0 = xv[0], p1 = xv[1], p2 = xv[4], p3 = xv[5];
    v16bf a;
#pragma unroll
    for (int i = 0; i < 4; ++i) {
      a[i]      = (__bf16)p0[i];
      a[i + 4]  = (__bf16)p1[i];
      a[i + 8]  = (__bf16)p2[i];
      a[i + 12] = (__bf16)p3[i];
    }
    v16bf b = bfrag[step * 32 + lane];            // 32B contiguous LDS read
    acc = __builtin_amdgcn_wmma_f32_16x16x32_bf16(false, a, false, b,
                                                  (short)0, acc, false, false);
  }

  // C/D: lane n = m16, VGPR r -> M = row0 + r + 8*half
  float* hp = h + (size_t)(row0 + 8 * half) * HID + m16;
  if (row0 + 16 <= nrows) {                       // uniform fast path: no per-store branches
#pragma unroll
    for (int r = 0; r < 8; ++r) hp[r * HID] = acc[r];
  } else {
#pragma unroll
    for (int r = 0; r < 8; ++r)
      if (row0 + 8 * half + r < nrows) hp[r * HID] = acc[r];
  }
}

// ---------------- layer-1 aggregation ----------------
__global__ __launch_bounds__(256) void k_self1(const float* __restrict__ h,
                                               const float* __restrict__ dinv,
                                               float* __restrict__ agg1, int nh) {
  int i = blockIdx.x * 256 + threadIdx.x;         // over n*HID
  if (i < nh) {
    float d = dinv[i >> 4];
    agg1[i] = d * d * h[i];                       // self-loop: norm = dinv^2
  }
}

__global__ __launch_bounds__(256) void k_edge1(const int* __restrict__ row,
                                               const int* __restrict__ col,
                                               const float* __restrict__ dinv,
                                               const float* __restrict__ h,
                                               float* __restrict__ agg1, int e) {
  int idx = blockIdx.x * 256 + threadIdx.x;       // 16 threads per edge
  int ei = idx >> 4, t = idx & 15;
  if (ei >= e) return;
  int r = row[ei], c = col[ei];
  float w = dinv[r] * dinv[c];
  atomicAdd(&agg1[(size_t)c * HID + t], w * h[(size_t)r * HID + t]);
}

// ---------------- relu + bias + layer-2 GEMM + self-loop init ----------------
__global__ __launch_bounds__(256) void k_layer2(const float* __restrict__ agg1,
                                                const float* __restrict__ b1,
                                                const float* __restrict__ W2,
                                                const float* __restrict__ dinv,
                                                float* __restrict__ z,
                                                float* __restrict__ agg2, int n) {
  __shared__ float w2s[HID * NCLS];
  __shared__ float b1s[HID];
  for (int i = threadIdx.x; i < HID * NCLS; i += 256) w2s[i] = W2[i];
  if (threadIdx.x < HID) b1s[threadIdx.x] = b1[threadIdx.x];
  __syncthreads();

  int node = blockIdx.x * 256 + threadIdx.x;
  if (node >= n) return;
  float hv[HID];
#pragma unroll
  for (int k = 0; k < HID; ++k)
    hv[k] = fmaxf(agg1[(size_t)node * HID + k] + b1s[k], 0.0f);
  float d = dinv[node], d2 = d * d;
  float* zp = z    + (size_t)node * NCLS;
  float* ap = agg2 + (size_t)node * NCLS;
#pragma unroll 4
  for (int c = 0; c < NCLS; ++c) {
    float s = 0.0f;
#pragma unroll
    for (int k = 0; k < HID; ++k) s = fmaf(hv[k], w2s[k * NCLS + c], s);
    zp[c] = s;
    ap[c] = d2 * s;                               // self-loop contribution
  }
}

__global__ __launch_bounds__(256) void k_edge2(const int* __restrict__ row,
                                               const int* __restrict__ col,
                                               const float* __restrict__ dinv,
                                               const float* __restrict__ z,
                                               float* __restrict__ agg2, int e) {
  int idx = blockIdx.x * 256 + threadIdx.x;       // 8 threads/edge x 5 feats
  int ei = idx >> 3, t = idx & 7;
  if (ei >= e) return;
  int r = row[ei], c = col[ei];
  float w = dinv[r] * dinv[c];
  const float* zp = z + (size_t)r * NCLS + t * 5;
  float* ap = agg2 + (size_t)c * NCLS + t * 5;
#pragma unroll
  for (int j = 0; j < 5; ++j) atomicAdd(&ap[j], w * zp[j]);
}

// ---------------- bias + log_softmax ----------------
__global__ __launch_bounds__(256) void k_logsoftmax(const float* __restrict__ agg2,
                                                    const float* __restrict__ b2,
                                                    float* __restrict__ out, int n) {
  __shared__ float b2s[NCLS];
  if (threadIdx.x < NCLS) b2s[threadIdx.x] = b2[threadIdx.x];
  __syncthreads();
  int node = blockIdx.x * 256 + threadIdx.x;
  if (node >= n) return;
  const float* ap = agg2 + (size_t)node * NCLS;
  float v[NCLS];
  float m = -3.402823466e38f;
#pragma unroll
  for (int c = 0; c < NCLS; ++c) { v[c] = ap[c] + b2s[c]; m = fmaxf(m, v[c]); }
  float s = 0.0f;
#pragma unroll
  for (int c = 0; c < NCLS; ++c) s += __expf(v[c] - m);
  float l = m + __logf(s);
  float* op = out + (size_t)node * NCLS;
#pragma unroll
  for (int c = 0; c < NCLS; ++c) op[c] = v[c] - l;
}

// ---------------- host launcher ----------------
extern "C" void kernel_launch(void* const* d_in, const int* in_sizes, int n_in,
                              void* d_out, int out_size, void* d_ws, size_t ws_size,
                              hipStream_t stream) {
  const float* x  = (const float*)d_in[0];
  const int*   ei = (const int*)  d_in[1];       // (2, E) int32
  const float* W1 = (const float*)d_in[2];
  const float* b1 = (const float*)d_in[3];
  const float* W2 = (const float*)d_in[4];
  const float* b2 = (const float*)d_in[5];
  float* out = (float*)d_out;

  const int n = in_sizes[0] / F_IN;
  const int e = in_sizes[1] / 2;
  const int* row = ei;          // sources (gather)
  const int* col = ei + e;      // targets (scatter)

  float* f    = (float*)d_ws;
  float* deg  = f;
  float* dinv = deg  + n;
  float* h    = dinv + n;
  float* agg1 = h    + (size_t)n * HID;
  float* z    = agg1 + (size_t)n * HID;
  float* agg2 = z    + (size_t)n * NCLS;

  auto cdiv = [](long long a, long long b) { return (int)((a + b - 1) / b); };

  k_deg_init  <<<cdiv(n, 256), 256, 0, stream>>>(deg, n);
  k_deg_edges <<<cdiv(e, 256), 256, 0, stream>>>(col, deg, e);
  k_dinv      <<<cdiv(n, 256), 256, 0, stream>>>(deg, dinv, n);

  const int ntiles = cdiv(n, 16);
  k_gemm1_wmma<<<cdiv(ntiles, 8), 256, 0, stream>>>(x, W1, h, n, ntiles);

  k_self1     <<<cdiv((long long)n * HID, 256), 256, 0, stream>>>(h, dinv, agg1, n * HID);
  k_edge1     <<<cdiv((long long)e * 16, 256), 256, 0, stream>>>(row, col, dinv, h, agg1, e);
  k_layer2    <<<cdiv(n, 256), 256, 0, stream>>>(agg1, b1, W2, dinv, z, agg2, n);
  k_edge2     <<<cdiv((long long)e * 8, 256), 256, 0, stream>>>(row, col, dinv, z, agg2, e);
  k_logsoftmax<<<cdiv(n, 256), 256, 0, stream>>>(agg2, b2, out, n);
}
